// RoIHead_66967130079763
// MI455X (gfx1250) — matrix operand order repaired and checked by reading, not compiled, not run on hardware
//
#include <hip/hip_runtime.h>
#include <hip/hip_bf16.h>

// ---------------------------------------------------------------------------
// Problem constants
// ---------------------------------------------------------------------------
#define RR   512          // rois
#define CC   512          // channels
#define HH   50
#define WW   50
#define PP   7
#define DD   (CC * PP * PP)   // 25088
#define FCN  4096
#define NLOC (21 * 4)         // 84
#define NSCR 21

typedef __attribute__((ext_vector_type(16))) __bf16 bf16x16;
typedef __attribute__((ext_vector_type(8)))  float  f32x8;

union FragU {                 // 32 bytes: two LDS b128 loads -> one WMMA operand
    uint4   u[2];
    bf16x16 v;
};

// fp32 -> bf16 bits, round-to-nearest-even (inputs are finite)
static __device__ __forceinline__ unsigned short f2bf(float f) {
    unsigned int u = __builtin_bit_cast(unsigned int, f);
    u += 0x7FFFu + ((u >> 16) & 1u);
    return (unsigned short)(u >> 16);
}

// ---------------------------------------------------------------------------
// Kernel 1: RoIPool (torchvision semantics) -> bf16 activation matrix [512, 25088]
// One thread per output bin element. Feature map (5 MB) lives in L2.
// ---------------------------------------------------------------------------
__global__ __launch_bounds__(256)
void roi_pool_bf16(const float* __restrict__ x,      // [C,H,W]
                   const float* __restrict__ rois,   // [R,4] (y1,x1,y2,x2)
                   unsigned short* __restrict__ outA /* [R, D] bf16 */) {
    int e = blockIdx.x * 256 + threadIdx.x;
    if (e >= RR * CC * PP * PP) return;
    int r   = e / (CC * PP * PP);
    int rem = e % (CC * PP * PP);
    int c   = rem / (PP * PP);
    int b   = rem % (PP * PP);
    int ph  = b / PP, pw = b % PP;

    const float s = 0.0625f;
    float y1 = rintf(rois[r * 4 + 0] * s);
    float x1 = rintf(rois[r * 4 + 1] * s);
    float y2 = rintf(rois[r * 4 + 2] * s);
    float x2 = rintf(rois[r * 4 + 3] * s);
    float bw = fmaxf(x2 - x1 + 1.0f, 1.0f) * (1.0f / PP);
    float bh = fmaxf(y2 - y1 + 1.0f, 1.0f) * (1.0f / PP);

    int ws = (int)fminf(fmaxf(x1 + floorf((float)pw * bw),        0.0f), (float)WW);
    int we = (int)fminf(fmaxf(x1 + ceilf ((float)(pw + 1) * bw),  0.0f), (float)WW);
    int hs = (int)fminf(fmaxf(y1 + floorf((float)ph * bh),        0.0f), (float)HH);
    int he = (int)fminf(fmaxf(y1 + ceilf ((float)(ph + 1) * bh),  0.0f), (float)HH);

    const float* xp = x + (size_t)c * (HH * WW);
    float m = -INFINITY;
    for (int h = hs; h < he; ++h)
        for (int w = ws; w < we; ++w)
            m = fmaxf(m, xp[h * WW + w]);
    if (!(m > -INFINITY)) m = 0.0f;   // empty bin -> 0

    outA[(size_t)r * DD + (size_t)c * (PP * PP) + b] = f2bf(m);
}

// ---------------------------------------------------------------------------
// Kernel 2: C[512,N] = act( A_bf16[512,K] @ W_f32[K,N] + bias ), WMMA bf16.
// Block tile 128x128, 8 waves (2 M x 4 N), wave tile 64x32 (8 accumulators),
// K-step 32. W converted fp32->bf16 in registers, staged transposed in LDS.
// ---------------------------------------------------------------------------
#define BM 128
#define BN 128
#define BK 32
#define LDP 40   // padded LDS row stride (shorts) to break bank-stride conflicts

__global__ __launch_bounds__(256)
void gemm_bf16_wmma(const unsigned short* __restrict__ A,   // [512,K] bf16
                    const float* __restrict__ Wg,           // [K,N] fp32
                    const float* __restrict__ bias,         // [N]
                    unsigned short* __restrict__ outB,      // bf16 out (or null)
                    float* __restrict__ outF,               // fp32 out (or null)
                    int K, int N, int doRelu) {
    __shared__ unsigned short sA[BM][LDP];   // [m][k]
    __shared__ unsigned short sB[BN][LDP];   // transposed: [n][k]

    const int tid    = threadIdx.x;
    const int lane   = tid & 31;
    const int wave   = tid >> 5;
    const int waveM  = (wave & 1) * 64;      // 2 waves along M
    const int waveN  = (wave >> 1) * 32;     // 4 waves along N
    const int mBlock = blockIdx.x * BM;
    const int nBlock = blockIdx.y * BN;
    const int l16    = lane & 15;
    const int khalf  = lane >> 4;

    f32x8 acc[4][2];
    const f32x8 zero = {0.f, 0.f, 0.f, 0.f, 0.f, 0.f, 0.f, 0.f};
#pragma unroll
    for (int i = 0; i < 4; ++i)
#pragma unroll
        for (int j = 0; j < 2; ++j) acc[i][j] = zero;

    // per-thread load coordinates
    const int ra = tid >> 1;                 // A: row 0..127
    const int ca = (tid & 1) * 16;           // A: k-half (16 bf16 = 32B)
    const int kb = tid >> 3;                 // B: k row 0..31
    const int nb = (tid & 7) * 16;           // B: 16 columns

    for (int k0 = 0; k0 < K; k0 += BK) {
        // ---- stage A tile (bf16, coalesced b128 pairs) ----
        {
            const uint4* src = reinterpret_cast<const uint4*>(
                A + (size_t)(mBlock + ra) * K + (k0 + ca));
            uint4 a0 = src[0];
            uint4 a1 = src[1];
            *reinterpret_cast<uint4*>(&sA[ra][ca])     = a0;
            *reinterpret_cast<uint4*>(&sA[ra][ca + 8]) = a1;
        }
        // ---- stage B tile: fp32 load, cvt bf16, transposed store ----
        {
            const float* wrow = Wg + (size_t)(k0 + kb) * N;
            __builtin_prefetch(wrow + (size_t)BK * N, 0, 0);  // next K-slab
            if (nBlock + BN <= N) {
                const float4* wv =
                    reinterpret_cast<const float4*>(wrow + nBlock + nb);
#pragma unroll
                for (int i = 0; i < 4; ++i) {
                    float4 f4 = wv[i];
                    sB[nb + 4 * i + 0][kb] = f2bf(f4.x);
                    sB[nb + 4 * i + 1][kb] = f2bf(f4.y);
                    sB[nb + 4 * i + 2][kb] = f2bf(f4.z);
                    sB[nb + 4 * i + 3][kb] = f2bf(f4.w);
                }
            } else {
#pragma unroll
                for (int i = 0; i < 16; ++i) {
                    int col = nBlock + nb + i;
                    sB[nb + i][kb] = f2bf(col < N ? wrow[col] : 0.0f);
                }
            }
        }
        __syncthreads();

        // ---- fragments from LDS (layouts per CDNA5 ISA 7.12.2) ----
        FragU fa[4], fb[2];
#pragma unroll
        for (int ms = 0; ms < 4; ++ms) {
            int m = waveM + ms * 16 + l16;
            fa[ms].u[0] = *reinterpret_cast<const uint4*>(&sA[m][khalf * 8]);
            fa[ms].u[1] = *reinterpret_cast<const uint4*>(&sA[m][16 + khalf * 8]);
        }
#pragma unroll
        for (int ns = 0; ns < 2; ++ns) {
            int n = waveN + ns * 16 + l16;
            fb[ns].u[0] = *reinterpret_cast<const uint4*>(&sB[n][khalf * 8]);
            fb[ns].u[1] = *reinterpret_cast<const uint4*>(&sB[n][16 + khalf * 8]);
        }

#pragma unroll
        for (int ms = 0; ms < 4; ++ms)
#pragma unroll
            for (int ns = 0; ns < 2; ++ns)
                acc[ms][ns] = __builtin_amdgcn_wmma_f32_16x16x32_bf16(
                    false, fa[ms].v, false, fb[ns].v,
                    (short)0, acc[ms][ns], false, false);

        __syncthreads();
    }

    // ---- epilogue: bias (+ReLU), store bf16 or fp32 ----
#pragma unroll
    for (int ms = 0; ms < 4; ++ms) {
#pragma unroll
        for (int ns = 0; ns < 2; ++ns) {
            int ng = nBlock + waveN + ns * 16 + l16;
            if (ng >= N) continue;
            float bv = bias[ng];
            f32x8 c = acc[ms][ns];
#pragma unroll
            for (int v = 0; v < 8; ++v) {
                int mg = mBlock + waveM + ms * 16 + v + 8 * khalf;
                float val = c[v] + bv;
                if (doRelu) val = fmaxf(val, 0.0f);
                size_t idx = (size_t)mg * N + ng;
                if (outB) outB[idx] = f2bf(val);
                else      outF[idx] = val;
            }
        }
    }
}

// ---------------------------------------------------------------------------
// Launch: pool -> fc1(relu) -> fc2(relu) -> cls_loc & score heads
// ---------------------------------------------------------------------------
extern "C" void kernel_launch(void* const* d_in, const int* in_sizes, int n_in,
                              void* d_out, int out_size, void* d_ws, size_t ws_size,
                              hipStream_t stream) {
    const float* x    = (const float*)d_in[0];
    const float* rois = (const float*)d_in[1];
    // d_in[2] roi_indices: batch N==1, all zero -> unused
    const float* W1 = (const float*)d_in[3];
    const float* b1 = (const float*)d_in[4];
    const float* W2 = (const float*)d_in[5];
    const float* b2 = (const float*)d_in[6];
    const float* W3 = (const float*)d_in[7];
    const float* b3 = (const float*)d_in[8];
    const float* W4 = (const float*)d_in[9];
    const float* b4 = (const float*)d_in[10];
    float* out = (float*)d_out;

    unsigned short* A1 = (unsigned short*)d_ws;                 // [512, 25088] bf16
    unsigned short* F6 = A1 + (size_t)RR * DD;                  // [512, 4096] bf16
    unsigned short* F7 = F6 + (size_t)RR * FCN;                 // [512, 4096] bf16

    // 1) RoI max-pool, emit bf16 activations
    roi_pool_bf16<<<(RR * CC * PP * PP) / 256, 256, 0, stream>>>(x, rois, A1);

    // 2) fc6 = relu(A1 @ W1 + b1)      [512 x 4096], K = 25088
    gemm_bf16_wmma<<<dim3(RR / BM, FCN / BN), 256, 0, stream>>>(
        A1, W1, b1, F6, nullptr, DD, FCN, 1);

    // 3) fc7 = relu(fc6 @ W2 + b2)     [512 x 4096], K = 4096
    gemm_bf16_wmma<<<dim3(RR / BM, FCN / BN), 256, 0, stream>>>(
        F6, W2, b2, F7, nullptr, FCN, FCN, 1);

    // 4) roi_cls_locs = fc7 @ W3 + b3  [512 x 84] fp32 -> d_out[0:]
    gemm_bf16_wmma<<<dim3(RR / BM, 1), 256, 0, stream>>>(
        F7, W3, b3, nullptr, out, FCN, NLOC, 0);

    // 5) roi_scores = fc7 @ W4 + b4    [512 x 21] fp32 -> d_out[512*84:]
    gemm_bf16_wmma<<<dim3(RR / BM, 1), 256, 0, stream>>>(
        F7, W4, b4, nullptr, out + (size_t)RR * NLOC, FCN, NSCR, 0);
}
